// VariationalAutoEncoder_8993661518175
// MI455X (gfx1250) — matrix operand order repaired and checked by reading, not compiled, not run on hardware
//
#include <hip/hip_runtime.h>
#include <hip/hip_bf16.h>

// ---------------------------------------------------------------------------
// Graph-VAE forward for MI455X (gfx1250, wave32).
// All GEMMs run through v_wmma_f32_16x16x32_bf16 (bf16 in, f32 accumulate)
// with fused bias / leaky-relu / batchnorm / relu epilogues.
// LDS tiles are stored in WMMA *fragment order* so each lane loads fragments
// as 32-byte vectors (2x ds_load_b128). Macro tile 128x64: each wave owns a
// 32x32 output patch -> 4 WMMAs per K-chunk with 2x register reuse of A/B.
// ---------------------------------------------------------------------------

#define N_NODES   65536
#define N_EDGES   524288
#define BATCH_B   512
#define D_IN      128
#define H_ENC     256
#define LAT       64
#define H_DEC     512
#define NMAX      128
#define NP        8128        // 128*127/2
#define NSTATS    7
#define BN_EPS    1e-5f

typedef __attribute__((ext_vector_type(16))) __bf16 v16bf;
typedef __attribute__((ext_vector_type(8)))  float  v8f;

__device__ __forceinline__ unsigned short f2bf(float f) {
    // round-to-nearest-even f32 -> bf16
    unsigned int u = __float_as_uint(f);
    u += 0x7FFFu + ((u >> 16) & 1u);
    return (unsigned short)(u >> 16);
}

// Epilogue modes
#define EPI_NONE     0
#define EPI_LRELU    1
#define EPI_LRELU_BN 2
#define EPI_RELU     3

// ---------------------------------------------------------------------------
// WMMA GEMM:  C[M x N] = epi(A[M x K] @ B[K x N] + bias)
// 256 threads = 8 waves; macro tile 128(M) x 64(N); K chunked by 32, double-
// buffered LDS in fragment order:
//   Afrag[mtile(8)][lane(32)][e(16)] ushort   (8 KB / buffer)
//   Bfrag[ntile(4)][lane(32)][e(16)] ushort   (4 KB / buffer)
// A fragment inverse map: lane = (m&15) + 16*hf, elements 0..7 <-> k = 8*hf..,
// elements 8..15 <-> k = 16+8*hf.. (contiguous k groups -> vector loads).
// B fragment inverse map: lane = (n&15) + 16*((k>>3)&1), element e = (k&7) + 8*(k>>4) (same k order as A).
// M must be a multiple of 128, N of 64 (true for every GEMM here); K is
// arbitrary (zero-padded tail handled in the staging path).
// ---------------------------------------------------------------------------
__global__ __launch_bounds__(256)
void wmma_gemm(const float* __restrict__ A, const float* __restrict__ B,
               const float* __restrict__ bias, float* __restrict__ C,
               int M, int K, int N, int ldc, int epi,
               const float* __restrict__ bn_g, const float* __restrict__ bn_be)
{
    __shared__ unsigned short Afrag[2][8 * 32 * 16];
    __shared__ unsigned short Bfrag[2][4 * 32 * 16];

    const int tid  = threadIdx.x;
    const int wave = tid >> 5;
    const int lane = tid & 31;
    const int half = lane >> 4;
    const int l15  = lane & 15;
    const int m0   = blockIdx.y * 128;
    const int n0   = blockIdx.x * 64;
    const int mtp  = (wave & 3) * 2;        // wave's first 16-row tile (0,2,4,6)
    const int nt0  = (wave >> 2) * 2;       // wave's first 16-col tile (0 or 2)

    // ---- staging thread coords ----
    // A: thread t == slot in [0,256): mtile = t>>5, fragment lane = t&31;
    //    stages both element groups g=0,1 (8 contiguous k each).
    const int a_mt   = tid >> 5;
    const int a_ln   = tid & 31;
    const int a_hf   = a_ln >> 4;
    const int a_row  = m0 + a_mt * 16 + (a_ln & 15);
    // B: thread t -> k = t>>3 in [0,32), ngroup = t&7 covers 8 consecutive n.
    const int b_k    = tid >> 3;
    const int b_ng   = tid & 7;
    const int b_hf   = (b_k >> 3) & 1;
    const int b_e    = (b_k & 7) + 8 * (b_k >> 4);

    const bool k32 = (K & 31) == 0;          // uniform fast-path flag

    v8f acc[4] = {v8f{}, v8f{}, v8f{}, v8f{}};   // [mi*2 + ni]

    auto stage = [&](int buf, int k0) {
        // ---------- A tile (128 x 32) ----------
        #pragma unroll
        for (int g = 0; g < 2; ++g) {
            const int cb = k0 + g * 16 + 8 * a_hf;
            const float* rp = A + (long)a_row * K + cb;
            unsigned short u[8];
            if (k32) {
                // 32B-aligned groups of 8 floats, always in range
                const float4* q = (const float4*)rp;
                float4 x0 = q[0], x1 = q[1];
                u[0] = f2bf(x0.x); u[1] = f2bf(x0.y);
                u[2] = f2bf(x0.z); u[3] = f2bf(x0.w);
                u[4] = f2bf(x1.x); u[5] = f2bf(x1.y);
                u[6] = f2bf(x1.z); u[7] = f2bf(x1.w);
            } else if (cb + 8 <= K) {
                #pragma unroll
                for (int j = 0; j < 8; ++j) u[j] = f2bf(rp[j]);
            } else {
                #pragma unroll
                for (int j = 0; j < 8; ++j)
                    u[j] = f2bf((cb + j < K) ? rp[j] : 0.0f);
            }
            uint4 pk;
            pk.x = (unsigned)u[0] | ((unsigned)u[1] << 16);
            pk.y = (unsigned)u[2] | ((unsigned)u[3] << 16);
            pk.z = (unsigned)u[4] | ((unsigned)u[5] << 16);
            pk.w = (unsigned)u[6] | ((unsigned)u[7] << 16);
            *(uint4*)&Afrag[buf][tid * 16 + g * 8] = pk;     // ds_store_b128
        }
        // ---------- B tile (32 x 64) ----------
        {
            const int krow = k0 + b_k;
            const float* bp = B + (long)krow * N + (n0 + b_ng * 8);
            float f[8];
            if (krow < K) {
                const float4* q = (const float4*)bp;   // N multiple of 8 here
                float4 x0 = q[0], x1 = q[1];
                f[0] = x0.x; f[1] = x0.y; f[2] = x0.z; f[3] = x0.w;
                f[4] = x1.x; f[5] = x1.y; f[6] = x1.z; f[7] = x1.w;
            } else {
                #pragma unroll
                for (int j = 0; j < 8; ++j) f[j] = 0.0f;
            }
            #pragma unroll
            for (int j = 0; j < 8; ++j) {
                int n  = b_ng * 8 + j;                  // 0..63
                int ln = (n & 15) + 16 * b_hf;
                Bfrag[buf][((n >> 4) * 32 + ln) * 16 + b_e] = f2bf(f[j]);
            }
        }
    };

    const int nchunks = (K + 31) >> 5;
    stage(0, 0);
    for (int c = 0; c < nchunks; ++c) {
        __syncthreads();                     // staged chunk c visible
        if (c + 1 < nchunks) stage((c + 1) & 1, (c + 1) * 32);
        const int buf = c & 1;
        v16bf a0 = *(const v16bf*)&Afrag[buf][((mtp + 0) * 32 + lane) * 16];
        v16bf a1 = *(const v16bf*)&Afrag[buf][((mtp + 1) * 32 + lane) * 16];
        v16bf b0 = *(const v16bf*)&Bfrag[buf][((nt0 + 0) * 32 + lane) * 16];
        v16bf b1 = *(const v16bf*)&Bfrag[buf][((nt0 + 1) * 32 + lane) * 16];
        acc[0] = __builtin_amdgcn_wmma_f32_16x16x32_bf16(false, a0, false, b0,
                                                         (short)0, acc[0], false, false);
        acc[1] = __builtin_amdgcn_wmma_f32_16x16x32_bf16(false, a0, false, b1,
                                                         (short)0, acc[1], false, false);
        acc[2] = __builtin_amdgcn_wmma_f32_16x16x32_bf16(false, a1, false, b0,
                                                         (short)0, acc[2], false, false);
        acc[3] = __builtin_amdgcn_wmma_f32_16x16x32_bf16(false, a1, false, b1,
                                                         (short)0, acc[3], false, false);
    }

    // --- epilogue + store (C/D layout: vgpr e -> m = e + 8*half, n = lane&15)
    const int nA = n0 + nt0 * 16 + l15;
    const int nB = nA + 16;
    float biasA = bias ? bias[nA] : 0.0f;
    float biasB = bias ? bias[nB] : 0.0f;
    float gA = 1.0f, beA = 0.0f, gB = 1.0f, beB = 0.0f;
    if (epi == EPI_LRELU_BN) {
        float s = rsqrtf(1.0f + BN_EPS);
        gA = bn_g[nA] * s; beA = bn_be[nA];
        gB = bn_g[nB] * s; beB = bn_be[nB];
    }
    #pragma unroll
    for (int mi = 0; mi < 2; ++mi) {
        const int mbase = m0 + (mtp + mi) * 16 + (half ? 8 : 0);
        #pragma unroll
        for (int e = 0; e < 8; ++e) {
            int m = mbase + e;
            float vA = acc[mi * 2 + 0][e] + biasA;
            float vB = acc[mi * 2 + 1][e] + biasB;
            if (epi == EPI_LRELU || epi == EPI_LRELU_BN) {
                vA = vA > 0.0f ? vA : 0.2f * vA;
                vB = vB > 0.0f ? vB : 0.2f * vB;
            }
            if (epi == EPI_LRELU_BN) { vA = vA * gA + beA; vB = vB * gB + beB; }
            if (epi == EPI_RELU) { vA = fmaxf(vA, 0.0f); vB = fmaxf(vB, 0.0f); }
            C[(long)m * ldc + nA] = vA;
            C[(long)m * ldc + nB] = vB;
        }
    }
}

// ---------------------------------------------------------------------------
// agg init: agg = h  (vectorized copy)
// ---------------------------------------------------------------------------
__global__ void copy_f4(const float4* __restrict__ src, float4* __restrict__ dst,
                        long n4)
{
    long i = (long)blockIdx.x * blockDim.x + threadIdx.x;
    if (i < n4) dst[i] = src[i];
}

// ---------------------------------------------------------------------------
// edge scatter: agg[dst] += h[src]   (one thread = one edge x 4 features)
// ---------------------------------------------------------------------------
__global__ void edge_scatter(const float* __restrict__ h, float* __restrict__ agg,
                             const int* __restrict__ src, const int* __restrict__ dst,
                             int D)
{
    long idx = (long)blockIdx.x * blockDim.x + threadIdx.x;
    int chunks = D >> 2;
    long total = (long)N_EDGES * chunks;
    if (idx >= total) return;
    int e = (int)(idx / chunks);
    int c = (int)(idx % chunks) * 4;
    int s = src[e], d = dst[e];
    const float* hp = h + (long)s * D + c;
    float* ap = agg + (long)d * D + c;
    atomicAdd(ap + 0, hp[0]);
    atomicAdd(ap + 1, hp[1]);
    atomicAdd(ap + 2, hp[2]);
    atomicAdd(ap + 3, hp[3]);
}

// ---------------------------------------------------------------------------
// pooled = segment_sum over 128 consecutive rows; concat stats; apply bn_eval.
// ---------------------------------------------------------------------------
__global__ __launch_bounds__(288)
void pool_concat_bn(const float* __restrict__ h, const float* __restrict__ stats,
                    const float* __restrict__ bn_g, const float* __restrict__ bn_b,
                    float* __restrict__ xin)
{
    int b = blockIdx.x;
    int c = threadIdx.x;
    if (c >= H_ENC + NSTATS) return;
    float v;
    if (c < H_ENC) {
        const float* hp = h + (long)b * NMAX * H_ENC + c;
        float s = 0.0f;
        #pragma unroll 4
        for (int i = 0; i < NMAX; ++i) s += hp[(long)i * H_ENC];
        v = s;
    } else {
        v = stats[b * NSTATS + (c - H_ENC)];
    }
    float sc = rsqrtf(1.0f + BN_EPS);
    xin[(long)b * (H_ENC + NSTATS) + c] = v * bn_g[c] * sc + bn_b[c];
}

// fill z[:, 64:71] = stats  (mu was GEMM'd into z[:, 0:64] with ldc=71)
__global__ void z_fill_stats(const float* __restrict__ stats, float* __restrict__ z)
{
    int b = blockIdx.x;
    int t = threadIdx.x;
    if (t < NSTATS) z[(long)b * (LAT + NSTATS) + LAT + t] = stats[b * NSTATS + t];
}

// ---------------------------------------------------------------------------
// gumbel straight-through is numerically `hard`: adj[b,i,j] = (y0 >= y1)
// for the upper-tri pair p(i,j); symmetric; diagonal 0.
// ---------------------------------------------------------------------------
__global__ void adj_build(const float* __restrict__ logits,
                          const float* __restrict__ gumbel,
                          float* __restrict__ adj)
{
    long idx = (long)blockIdx.x * blockDim.x + threadIdx.x;
    if (idx >= (long)BATCH_B * NMAX * NMAX) return;
    int b = (int)(idx >> 14);
    int r = (int)(idx >> 7) & (NMAX - 1);
    int c = (int)idx & (NMAX - 1);
    float v = 0.0f;
    if (r != c) {
        int i = r < c ? r : c;
        int j = r < c ? c : r;
        int p = i * (NMAX - 1) - (i * (i - 1)) / 2 + (j - i - 1);
        const float* lg = logits + (long)b * (2 * NP) + 2 * p;
        const float* gu = gumbel + ((long)b * NP + p) * 2;
        float y0 = lg[0] + gu[0];
        float y1 = lg[1] + gu[1];
        v = (y0 >= y1) ? 1.0f : 0.0f;   // argmax tie -> index 0
    }
    adj[idx] = v;
}

// ---------------------------------------------------------------------------
// host-side orchestration
// ---------------------------------------------------------------------------
static inline void launch_gemm(const float* A, const float* B, const float* bias,
                               float* C, int M, int K, int N, int ldc, int epi,
                               const float* g, const float* be, hipStream_t s)
{
    dim3 grid(N / 64, M / 128);
    wmma_gemm<<<grid, 256, 0, s>>>(A, B, bias, C, M, K, N, ldc, epi, g, be);
}

extern "C" void kernel_launch(void* const* d_in, const int* in_sizes, int n_in,
                              void* d_out, int out_size, void* d_ws, size_t ws_size,
                              hipStream_t stream)
{
    (void)in_sizes; (void)n_in; (void)out_size; (void)ws_size;

    const float* x      = (const float*)d_in[0];
    const int*   eidx   = (const int*)  d_in[1];   // [2, E]
    /* d_in[2] = batch: arange//128, structure exploited directly */
    const float* stats  = (const float*)d_in[3];
    const float* gumbel = (const float*)d_in[4];

    struct Conv { const float *w1, *b1, *g, *be, *w2, *b2; } L[3];
    for (int l = 0; l < 3; ++l) {
        int base = 5 + 6 * l;
        L[l].w1 = (const float*)d_in[base + 0];
        L[l].b1 = (const float*)d_in[base + 1];
        L[l].g  = (const float*)d_in[base + 2];
        L[l].be = (const float*)d_in[base + 3];
        L[l].w2 = (const float*)d_in[base + 4];
        L[l].b2 = (const float*)d_in[base + 5];
    }
    const float* bn_g = (const float*)d_in[23];
    const float* bn_b = (const float*)d_in[24];
    const float* fc_w = (const float*)d_in[25];
    const float* fc_b = (const float*)d_in[26];
    const float* mu_w = (const float*)d_in[27];
    const float* mu_b = (const float*)d_in[28];
    const float* d0_w = (const float*)d_in[29];
    const float* d0_b = (const float*)d_in[30];
    const float* d1_w = (const float*)d_in[31];
    const float* d1_b = (const float*)d_in[32];
    const float* d2_w = (const float*)d_in[33];
    const float* d2_b = (const float*)d_in[34];

    const int* src = eidx;
    const int* dst = eidx + N_EDGES;

    // workspace: two 67MB ping-pong activation buffers; decoder smalls reuse
    // the freed one after the GNN stack.
    const size_t S = (size_t)N_NODES * H_ENC * sizeof(float);
    float* big0 = (float*)d_ws;
    float* big1 = (float*)((char*)d_ws + S);

    // ---------------- GNN stack ----------------
    const float* hp = x;
    float* p0 = big0;
    float* p1 = big1;
    int din = D_IN;
    for (int l = 0; l < 3; ++l) {
        long n4 = (long)N_NODES * din / 4;
        copy_f4<<<(unsigned)((n4 + 255) / 256), 256, 0, stream>>>(
            (const float4*)hp, (float4*)p0, n4);
        long tot = (long)N_EDGES * (din / 4);
        edge_scatter<<<(unsigned)((tot + 255) / 256), 256, 0, stream>>>(
            hp, p0, src, dst, din);
        // t = bn(lrelu(agg @ w1 + b1))
        launch_gemm(p0, L[l].w1, L[l].b1, p1, N_NODES, din, H_ENC, H_ENC,
                    EPI_LRELU_BN, L[l].g, L[l].be, stream);
        // h = lrelu(t @ w2 + b2)
        launch_gemm(p1, L[l].w2, L[l].b2, p0, N_NODES, H_ENC, H_ENC, H_ENC,
                    EPI_LRELU, nullptr, nullptr, stream);
        hp = p0;
        din = H_ENC;
        float* t = p0; p0 = p1; p1 = t;
    }
    // final h lives at big0; big1 free for decoder scratch

    float* xin    = big1;                         // 512 x 263
    float* fcout  = big1 + 135168;                // 512 x 256
    float* z      = big1 + 266240;                // 512 x 71
    float* dbuf0  = big1 + 302592;                // 512 x 512
    float* dbuf1  = big1 + 564736;                // 512 x 512
    float* logits = big1 + 826880;                // 512 x 16256

    // ---------------- head ----------------
    pool_concat_bn<<<BATCH_B, 288, 0, stream>>>(hp, stats, bn_g, bn_b, xin);

    launch_gemm(xin, fc_w, fc_b, fcout, BATCH_B, H_ENC + NSTATS, H_ENC, H_ENC,
                EPI_NONE, nullptr, nullptr, stream);
    // mu written directly into z with ldc = 71
    launch_gemm(fcout, mu_w, mu_b, z, BATCH_B, H_ENC, LAT, LAT + NSTATS,
                EPI_NONE, nullptr, nullptr, stream);
    z_fill_stats<<<BATCH_B, 32, 0, stream>>>(stats, z);

    launch_gemm(z, d0_w, d0_b, dbuf0, BATCH_B, LAT + NSTATS, H_DEC, H_DEC,
                EPI_RELU, nullptr, nullptr, stream);
    launch_gemm(dbuf0, d1_w, d1_b, dbuf1, BATCH_B, H_DEC, H_DEC, H_DEC,
                EPI_RELU, nullptr, nullptr, stream);
    launch_gemm(dbuf1, d2_w, d2_b, logits, BATCH_B, H_DEC, 2 * NP, 2 * NP,
                EPI_NONE, nullptr, nullptr, stream);

    long adj_n = (long)BATCH_B * NMAX * NMAX;
    adj_build<<<(unsigned)((adj_n + 255) / 256), 256, 0, stream>>>(
        logits, gumbel, (float*)d_out);
}
